// RnnEncoder_35648228556888
// MI455X (gfx1250) — matrix operand order repaired and checked by reading, not compiled, not run on hardware
//
#include <hip/hip_runtime.h>
#include <hip/hip_bf16.h>

typedef __attribute__((ext_vector_type(16))) __bf16 v16bf;
typedef __attribute__((ext_vector_type(8)))  __bf16 v8bf;
typedef __attribute__((ext_vector_type(8)))  float  v8f;
typedef __attribute__((ext_vector_type(4)))  unsigned int v4u;
typedef __attribute__((ext_vector_type(8)))  int v8i;
typedef __attribute__((ext_vector_type(4)))  int v4i;

#define WMMA_BF16(A,B,C) \
  __builtin_amdgcn_wmma_f32_16x16x32_bf16(false,(A),false,(B),(short)0,(C),false,false)

#define CAT16(lo,hi) \
  __builtin_shufflevector((lo),(hi),0,1,2,3,4,5,6,7,8,9,10,11,12,13,14,15)

#if defined(__has_builtin)
#if __has_builtin(__builtin_amdgcn_tensor_load_to_lds) && \
    __has_builtin(__builtin_amdgcn_s_wait_tensorcnt)
#define HAVE_TDM 1
#endif
#endif

__device__ __forceinline__ float sigmoidf_(float x) { return 1.0f / (1.0f + __expf(-x)); }

// ---------------------------------------------------------------------------
// fp32 -> bf16 conversion
// ---------------------------------------------------------------------------
__global__ void k_f2bf(const float* __restrict__ src, __bf16* __restrict__ dst, int n) {
  int i = blockIdx.x * blockDim.x + threadIdx.x;
  if (i < n) dst[i] = (__bf16)src[i];
}

// ---------------------------------------------------------------------------
// Input projection:  xg[ts][n][b] = sum_k x[b][ts][k]*W_ih[n][k] + b_ih[n]+b_hh[n]
// M = S*B = 32768 (row m: s=m>>5, b=m&31), N = 2048, K = 1024.
// 128x128 block tile, 8 waves, each wave 64x32 (4x2 WMMA tiles).
// B (weight) tile staged into LDS by the Tensor Data Mover; A staged manually.
// xg stored (S, 4H, B) so the scan kernel reads contiguous v8f per tile.
// ---------------------------------------------------------------------------
__global__ __launch_bounds__(256)
void k_xproj(const __bf16* __restrict__ xb,
             const __bf16* __restrict__ Wf, const __bf16* __restrict__ Wb,
             const float* __restrict__ bihf, const float* __restrict__ bhhf,
             const float* __restrict__ bihb, const float* __restrict__ bhhb,
             float* __restrict__ xgf, float* __restrict__ xgb)
{
  const int dir = blockIdx.z;
  const __bf16* __restrict__ W  = dir ? Wb   : Wf;
  const float*  __restrict__ bi = dir ? bihb : bihf;
  const float*  __restrict__ bh = dir ? bhhb : bhhf;
  float*        __restrict__ xg = dir ? xgb  : xgf;

  __shared__ __bf16 At[128 * 40];   // 128 M-rows, 32 k + 8 pad (16B)
  __shared__ __bf16 Bt[128 * 40];   // 128 N-rows, 32 k + 8 pad (16B)

  const int tid  = threadIdx.x;
  const int lane = tid & 31;
  const int w    = tid >> 5;
  const int half = lane >> 4;
  const int l16  = lane & 15;
  const int wm   = w >> 2;          // 0..1  (M quadrant, 64 rows)
  const int wn   = w & 3;           // 0..3  (N quadrant, 32 cols)
  const int m0   = blockIdx.x * 128;
  const int n0   = blockIdx.y * 128;

  v8f acc[4][2];
  #pragma unroll
  for (int mt = 0; mt < 4; ++mt)
    #pragma unroll
    for (int nt = 0; nt < 2; ++nt)
      #pragma unroll
      for (int r = 0; r < 8; ++r) acc[mt][nt][r] = 0.0f;

  const int ldr = tid >> 1;            // 0..127
  const int ldc = (tid & 1) * 16;      // 0 or 16
  const int mA  = m0 + ldr;            // global M row; s = mA>>5, b = mA&31
  const long aRow = ((long)(mA & 31) * 1024 + (mA >> 5)) * 1024;
#if !defined(HAVE_TDM)
  const long bRow = (long)(n0 + ldr) * 1024;
#endif

  for (int k0 = 0; k0 < 1024; k0 += 32) {
    v8bf a0 = *(const v8bf*)(xb + aRow + k0 + ldc);
    v8bf a1 = *(const v8bf*)(xb + aRow + k0 + ldc + 8);
#if !defined(HAVE_TDM)
    v8bf b0 = *(const v8bf*)(W  + bRow + k0 + ldc);
    v8bf b1 = *(const v8bf*)(W  + bRow + k0 + ldc + 8);
#endif
    __syncthreads();                               // prior-iter LDS reads done

#if defined(HAVE_TDM)
    if (tid < 32) {
      // TDM: 2D tile, 128 rows x 32 bf16, row stride 1024 elements, into Bt
      // with LDS padding of 4 DWORDs after every 16 DWORDs (matches 40-elem rows).
      unsigned long long ga =
          (unsigned long long)(const void*)(W + (long)n0 * 1024 + k0);
      unsigned int lds = (unsigned int)(unsigned long long)(void*)&Bt[0];
      v4u g0;
      g0[0] = 1u;                                          // count=1
      g0[1] = lds;                                         // lds_addr
      g0[2] = (unsigned int)(ga & 0xFFFFFFFFu);            // global_addr lo
      g0[3] = (unsigned int)((ga >> 32) & 0x1FFFFFFu)      // global_addr hi
              | (2u << 30);                                // type=2 (image)
      v8i g1;
      g1[0] = (1 << 16)       // data_size = 2 bytes
            | (1 << 20)       // pad_enable
            | (3 << 22)       // pad_interval: 16 DWORDs (=32 bf16 row)
            | (3 << 25);      // pad_amount:   4 DWORDs  (=16B pad)
      g1[1] = (1024 & 0xFFFF) << 16;                 // tensor_dim0 = 1024
      g1[2] = (2048 & 0xFFFF) << 16;                 // tensor_dim1 = 2048
      g1[3] = (32 & 0xFFFF) << 16;                   // tile_dim0 = 32
      g1[4] = 128;                                   // tile_dim1 = 128
      g1[5] = 1024;                                  // tensor_dim0_stride
      g1[6] = 0;
      g1[7] = 0;
      v4i gz4 = {0, 0, 0, 0};                        // groups 2/3: NULL (2D)
      v8i gz8 = {0, 0, 0, 0, 0, 0, 0, 0};
      __builtin_amdgcn_tensor_load_to_lds(g0, g1, gz4, gz4, gz8, 0);
    }
#endif
    // A tile staged manually (rows of the A tile are scattered in x: m=(s,b))
    *(v8bf*)(At + ldr * 40 + ldc)     = a0;
    *(v8bf*)(At + ldr * 40 + ldc + 8) = a1;
#if defined(HAVE_TDM)
    __builtin_amdgcn_s_wait_tensorcnt(0);
#else
    *(v8bf*)(Bt + ldr * 40 + ldc)     = b0;
    *(v8bf*)(Bt + ldr * 40 + ldc + 8) = b1;
#endif
    __syncthreads();

    v16bf afr[4], bfr[2];
    #pragma unroll
    for (int mt = 0; mt < 4; ++mt) {
      const __bf16* p = At + (wm * 64 + mt * 16 + l16) * 40;
      v8bf lo = *(const v8bf*)(p + half * 8);       // K = half*8 .. +7
      v8bf hi = *(const v8bf*)(p + 16 + half * 8);  // K = 16+half*8 .. +7
      afr[mt] = CAT16(lo, hi);
    }
    #pragma unroll
    for (int nt = 0; nt < 2; ++nt) {
      const __bf16* p = Bt + (wn * 32 + nt * 16 + l16) * 40;
      v8bf lo = *(const v8bf*)(p + half * 8);
      v8bf hi = *(const v8bf*)(p + 16 + half * 8);
      bfr[nt] = CAT16(lo, hi);
    }
    #pragma unroll
    for (int mt = 0; mt < 4; ++mt)
      #pragma unroll
      for (int nt = 0; nt < 2; ++nt)
        acc[mt][nt] = WMMA_BF16(afr[mt], bfr[nt], acc[mt][nt]);
  }

  // epilogue: + (b_ih + b_hh), store to xg in (S, 4H, B) layout
  #pragma unroll
  for (int nt = 0; nt < 2; ++nt) {
    const int n = n0 + wn * 32 + nt * 16 + l16;
    const float bs = bi[n] + bh[n];
    #pragma unroll
    for (int mt = 0; mt < 4; ++mt)
      #pragma unroll
      for (int r = 0; r < 8; ++r) {
        const int m = m0 + wm * 64 + mt * 16 + r + 8 * half;  // m = s*32+b
        xg[((long)(m >> 5) * 2048 + n) * 32 + (m & 31)] = acc[mt][nt][r] + bs;
      }
  }
}

// ---------------------------------------------------------------------------
// Recurrent scan. One 1024-thread block (32 waves) per direction.
// Wave w owns hidden columns [w*16, w*16+16); per step computes the four gate
// tiles (i,f,g,o) x 2 batch tiles with K=512 bf16 WMMA.  c in registers,
// h staged in LDS as bf16, W_hh streamed from L2 (with prefetch) every step.
// ---------------------------------------------------------------------------
__global__ __launch_bounds__(1024)
void k_lstm(const __bf16* __restrict__ Whhf, const __bf16* __restrict__ Whhb,
            const float* __restrict__ xgf,  const float* __restrict__ xgb,
            float* __restrict__ hidden)     // (B=32, S=1024, 2H=1024)
{
  const int dir = blockIdx.x;
  const __bf16* __restrict__ W  = dir ? Whhb : Whhf;
  const float*  __restrict__ xg = dir ? xgb  : xgf;

  __shared__ __bf16 hb[32 * 520];   // h (32 x 512) bf16, +8 pad per row

  const int tid  = threadIdx.x;
  const int lane = tid & 31;
  const int w    = tid >> 5;        // 0..31
  const int half = lane >> 4;
  const int l16  = lane & 15;
  const int jb   = w * 16;          // hidden column base for this wave

  for (int i = tid; i < 32 * 520; i += 1024) hb[i] = (__bf16)0.0f;
  __syncthreads();

  float c[16];
  #pragma unroll
  for (int i = 0; i < 16; ++i) c[i] = 0.0f;

  // per-lane W row base (lane = gate column), reused every step
  const __bf16* Wrow[4];
  #pragma unroll
  for (int g = 0; g < 4; ++g)
    Wrow[g] = W + (long)(g * 512 + jb + l16) * 512 + half * 16;

  for (int t = 0; t < 1024; ++t) {
    const int ts = dir ? (1023 - t) : t;

    // accumulators initialized from xg[ts] ((S,4H,B): 8 contiguous floats/lane)
    v8f acc[4][2];
    #pragma unroll
    for (int g = 0; g < 4; ++g)
      #pragma unroll
      for (int mt = 0; mt < 2; ++mt)
        acc[g][mt] = *(const v8f*)(xg + ((long)ts * 2048 + g * 512 + jb + l16) * 32
                                      + mt * 16 + 8 * half);

    // g += h_prev @ W_hh^T   (K = 512, 16 k-steps of 32)
    for (int kk = 0; kk < 16; ++kk) {
      const int kb = kk * 32;
      v16bf afr[2];
      #pragma unroll
      for (int mt = 0; mt < 2; ++mt) {
        const __bf16* p = hb + (mt * 16 + l16) * 520 + kb;
        v8bf lo = *(const v8bf*)(p + half * 16);
        v8bf hi = *(const v8bf*)(p + half * 16 + 8);
        afr[mt] = CAT16(lo, hi);
      }
      #pragma unroll
      for (int g = 0; g < 4; ++g) {
        // B fragment: lane = column n, 16 contiguous K values (32B load)
        v16bf bfr = *(const v16bf*)(Wrow[g] + kb);
        if (kk < 15) __builtin_prefetch(Wrow[g] + kb + 32, 0, 0);
        #pragma unroll
        for (int mt = 0; mt < 2; ++mt)
          acc[g][mt] = WMMA_BF16(afr[mt], bfr, acc[g][mt]);
      }
    }
    __syncthreads();   // all reads of hb for this step complete

    // elementwise gate update: i,f,g,o live in matching lanes -> no shuffle
    #pragma unroll
    for (int mt = 0; mt < 2; ++mt)
      #pragma unroll
      for (int r = 0; r < 8; ++r) {
        const int m  = mt * 16 + r + 8 * half;       // batch row
        const int j  = jb + l16;                     // hidden column
        const int ci = mt * 8 + r;
        const float iv = sigmoidf_(acc[0][mt][r]);
        const float fv = sigmoidf_(acc[1][mt][r]);
        const float gv = tanhf(acc[2][mt][r]);
        const float ov = sigmoidf_(acc[3][mt][r]);
        c[ci] = fv * c[ci] + iv * gv;
        const float h = ov * tanhf(c[ci]);
        hidden[((long)m * 1024 + ts) * 1024 + dir * 512 + j] = h;
        hb[m * 520 + j] = (__bf16)h;
      }
    __syncthreads();   // h visible to all waves for next step
  }
}

// ---------------------------------------------------------------------------
// Span gather: out[b][mm][k] ; k<512 -> h_fwd at end-1, k>=512 -> h_bwd at start
// ---------------------------------------------------------------------------
__global__ void k_gather(const long long* __restrict__ span,
                         const float* __restrict__ hidden,
                         float* __restrict__ out)
{
  const int idx = blockIdx.x * blockDim.x + threadIdx.x;
  if (idx >= 32 * 32 * 1024) return;
  const int k  = idx & 1023;
  const int mm = (idx >> 10) & 31;
  const int b  = idx >> 15;
  const long long s0 = span[((long)b * 32 + mm) * 2 + 0];
  const long long s1 = span[((long)b * 32 + mm) * 2 + 1];
  float v = 0.0f;
  if (s0 != -1) {
    int row = (k < 512) ? ((int)s1 - 1) : (int)s0;
    row = row < 0 ? 0 : (row > 1023 ? 1023 : row);
    v = hidden[((long)b * 1024 + row) * 1024 + k];
  }
  out[idx] = v;
}

// ---------------------------------------------------------------------------
extern "C" void kernel_launch(void* const* d_in, const int* in_sizes, int n_in,
                              void* d_out, int out_size, void* d_ws, size_t ws_size,
                              hipStream_t stream)
{
  const float* x    = (const float*)d_in[0];
  const float* Wihf = (const float*)d_in[1];
  const float* Whhf = (const float*)d_in[2];
  const float* bihf = (const float*)d_in[3];
  const float* bhhf = (const float*)d_in[4];
  const float* Wihb = (const float*)d_in[5];
  const float* Whhb = (const float*)d_in[6];
  const float* bihb = (const float*)d_in[7];
  const float* bhhb = (const float*)d_in[8];
  const long long* span = (const long long*)d_in[9];

  char* ws = (char*)d_ws;
  size_t off = 0;
  auto carve = [&](size_t bytes) -> char* {
    char* p = ws + off;
    off += (bytes + 255) & ~(size_t)255;
    return p;
  };
  __bf16* xb    = (__bf16*)carve(33554432ull * 2);  // x bf16
  __bf16* Wihfb = (__bf16*)carve(2097152ull * 2);
  __bf16* Wihbb = (__bf16*)carve(2097152ull * 2);
  __bf16* Whhfb = (__bf16*)carve(1048576ull * 2);
  __bf16* Whhbb = (__bf16*)carve(1048576ull * 2);
  float*  xgf   = (float*)carve(67108864ull * 4);   // (S, 4H, B)
  float*  xgb   = (float*)carve(67108864ull * 4);

  float* out    = (float*)d_out;
  float* hidden = out + 32 * 32 * 1024;             // hidden_all region

  const int thr = 256;
  k_f2bf<<<(33554432 + thr - 1) / thr, thr, 0, stream>>>(x,    xb,    33554432);
  k_f2bf<<<( 2097152 + thr - 1) / thr, thr, 0, stream>>>(Wihf, Wihfb,  2097152);
  k_f2bf<<<( 2097152 + thr - 1) / thr, thr, 0, stream>>>(Wihb, Wihbb,  2097152);
  k_f2bf<<<( 1048576 + thr - 1) / thr, thr, 0, stream>>>(Whhf, Whhfb,  1048576);
  k_f2bf<<<( 1048576 + thr - 1) / thr, thr, 0, stream>>>(Whhb, Whhbb,  1048576);

  dim3 gproj(256, 16, 2);   // (M/128, N/128, dir)
  k_xproj<<<gproj, 256, 0, stream>>>(xb, Wihfb, Wihbb, bihf, bhhf, bihb, bhhb, xgf, xgb);

  k_lstm<<<2, 1024, 0, stream>>>(Whhfb, Whhbb, xgf, xgb, hidden);

  k_gather<<<(1048576 + 255) / 256, 256, 0, stream>>>(span, hidden, out);
}
